// DisentangleEncoder_70248485093391
// MI455X (gfx1250) — matrix operand order repaired and checked by reading, not compiled, not run on hardware
//
#include <hip/hip_runtime.h>

#define NNODE 100000
#define NPAD  100032   // padded to 64-row blocks (1563 blocks)
#define NEDGE 1600000
#define FDIM  128
#define NFAC  4
#define DDIM  64
#define NLAY  3
#define GCNT  128

typedef __attribute__((ext_vector_type(16))) __bf16 v16bf;
typedef __attribute__((ext_vector_type(8)))  __bf16 v8bf;
typedef __attribute__((ext_vector_type(8)))  float  v8f;

// ---------------- WMMA fragment helpers (CDNA5 wave32 layouts) ----------------

__device__ __forceinline__ v8f wmma_bf16(v16bf a, v16bf b, v8f c) {
  return __builtin_amdgcn_wmma_f32_16x16x32_bf16(false, a, false, b, (short)0, c,
                                                 false, false);
}

// A fragment: 16x32 bf16 tile from row-major f32, p0 = row 0 of the 16-row tile.
// lanes 0-15: M=lane, K = {k0..k0+7} U {k0+16..k0+23}; lanes 16-31: +8 on K.
__device__ __forceinline__ v16bf loadA_f32(const float* __restrict__ p0, int lda,
                                           int lane, int k0) {
  const float* p = p0 + (lane & 15) * lda + k0 + (((lane >> 4) & 1) << 3);
  v16bf a;
#pragma unroll
  for (int i = 0; i < 8; ++i) {
    int k = 2 * i + ((i >= 4) ? 8 : 0);
    a[2 * i]     = (__bf16)p[k];
    a[2 * i + 1] = (__bf16)p[k + 1];
  }
  return a;
}

// A fragment from global bf16 (row-major, ld elems/row): two aligned 16B chunks
__device__ __forceinline__ v16bf loadA_gbf(const __bf16* __restrict__ p0, int lda,
                                           int lane, int k0) {
  const __bf16* p = p0 + (lane & 15) * lda + k0 + (((lane >> 4) & 1) << 3);
  v8bf lo = *(const v8bf*)p;         // K = base .. base+7
  v8bf hi = *(const v8bf*)(p + 16);  // K = base+16 .. base+23
  return __builtin_shufflevector(lo, hi, 0, 1, 2, 3, 4, 5, 6, 7,
                                 8, 9, 10, 11, 12, 13, 14, 15);
}

// Same pattern from LDS
__device__ __forceinline__ v16bf loadA_lds(const __bf16* m, int ld, int rbase,
                                           int lane, int k0) {
  const __bf16* p = m + (rbase + (lane & 15)) * ld + k0 + (((lane >> 4) & 1) << 3);
  v8bf lo = *(const v8bf*)p;
  v8bf hi = *(const v8bf*)(p + 16);
  return __builtin_shufflevector(lo, hi, 0, 1, 2, 3, 4, 5, 6, 7,
                                 8, 9, 10, 11, 12, 13, 14, 15);
}

// B fragment from pre-packed bf16 weights: one contiguous 32B load per lane.
// Packed layout: pk[(((nt*nKT + kt) * 32) + lane) * 16 + j]
__device__ __forceinline__ v16bf loadB_pk(const __bf16* __restrict__ pk, int nKT,
                                          int kt, int nt, int lane) {
  return *(const v16bf*)(pk + (((nt * nKT + kt) << 5) + lane) * 16);
}

// ---------------- one-time pre-pack / pre-convert kernels ----------------

// Pack row-major W[K][N] into B-fragment lane order.
__global__ void packB_kernel(const float* __restrict__ W, __bf16* __restrict__ pk,
                             int K, int N) {
  int idx = blockIdx.x * blockDim.x + threadIdx.x;
  if (idx >= K * N) return;
  int j    = idx & 15;
  int lane = (idx >> 4) & 31;
  int t    = idx >> 9;            // nt*nKT + kt
  int nKT  = K >> 5;
  int kt   = t % nKT;
  int nt   = t / nKT;
  int k = kt * 32 + (((lane >> 4) & 1) << 3) + (j & 7) + ((j >> 3) << 4);
  int n = nt * 16 + (lane & 15);
  pk[idx] = (__bf16)W[k * N + n];
}

// Pack transposed view: B[k][n] = W[n][k], W row-major [NB][K]
__global__ void packBT_kernel(const float* __restrict__ W, __bf16* __restrict__ pk,
                              int K, int NB) {
  int idx = blockIdx.x * blockDim.x + threadIdx.x;
  if (idx >= K * NB) return;
  int j    = idx & 15;
  int lane = (idx >> 4) & 31;
  int t    = idx >> 9;
  int nKT  = K >> 5;
  int kt   = t % nKT;
  int nt   = t / nKT;
  int k = kt * 32 + (((lane >> 4) & 1) << 3) + (j & 7) + ((j >> 3) << 4);
  int n = nt * 16 + (lane & 15);
  pk[idx] = (__bf16)W[n * K + k];
}

// f32 -> bf16 bulk convert (for x, reused by all 4 factors)
__global__ __launch_bounds__(256) void cvt_bf16_kernel(const float* __restrict__ s,
                                                       __bf16* __restrict__ d, long n) {
  long i = (long)blockIdx.x * blockDim.x + threadIdx.x;
  if (i < n) d[i] = (__bf16)s[i];
}

// ---------------- kernels ----------------

__global__ void zero_kernel(float* __restrict__ p, long n) {
  long i = (long)blockIdx.x * blockDim.x + threadIdx.x;
  if (i < n) p[i] = 0.0f;
}

__global__ void count_kernel(const int* __restrict__ batch, float* __restrict__ counts) {
  int i = blockIdx.x * blockDim.x + threadIdx.x;
  if (i < NNODE) atomicAdd(&counts[batch[i]], 1.0f);
}

// out = h = xb = x @ lin_W + lin_b ;  grid.x = N/16 row tiles, 4 waves = col tiles
__global__ __launch_bounds__(128) void lin_kernel(const __bf16* __restrict__ xbf,
                                                  const __bf16* __restrict__ pkW,
                                                  const float* __restrict__ bias,
                                                  float* __restrict__ out,
                                                  float* __restrict__ h,
                                                  __bf16* __restrict__ xb) {
  int lane = threadIdx.x & 31;
  int wave = threadIdx.x >> 5;         // 0..3 -> D col tile
  int rowbase = blockIdx.x * 16;
  const __bf16* xp = xbf + (long)rowbase * FDIM;
  v8f acc = {};
#pragma unroll
  for (int k0 = 0; k0 < FDIM; k0 += 32) {
    v16bf a = loadA_gbf(xp, FDIM, lane, k0);
    v16bf b = loadB_pk(pkW, FDIM / 32, k0 >> 5, wave, lane);
    acc = wmma_bf16(a, b, acc);
  }
  int col = wave * 16 + (lane & 15);
  float bv = bias[col];
  int roff = (lane < 16) ? 0 : 8;
#pragma unroll
  for (int r = 0; r < 8; ++r) {
    long idx = (long)(rowbase + r + roff) * DDIM + col;
    float v = acc[r] + bv;
    out[idx] = v;
    h[idx]   = v;
    xb[idx]  = (__bf16)v;
  }
}

// agg[dst] += xb[src] * w  (bf16 gather, f32 atomic accumulate)
__global__ __launch_bounds__(256) void scatter_kernel(const __bf16* __restrict__ xb,
                                                      const int* __restrict__ src,
                                                      const int* __restrict__ dst,
                                                      const float* __restrict__ w,
                                                      float* __restrict__ agg) {
  long idx = (long)blockIdx.x * blockDim.x + threadIdx.x;
  if (idx >= (long)NEDGE * DDIM) return;
  int e = (int)(idx >> 6);
  int d = (int)(idx & 63);
  float v = (float)xb[(long)src[e] * DDIM + d] * w[e];
  atomicAdd(&agg[(long)dst[e] * DDIM + d], v);
}

// Fused GraphConv + GRU step for one 64-row node tile. 8 waves.
__global__ __launch_bounds__(256) void layer_kernel(
    const float* __restrict__ agg, float* __restrict__ out, float* __restrict__ h,
    __bf16* __restrict__ xb,
    const __bf16* __restrict__ pkRel, const float* __restrict__ brel,
    const __bf16* __restrict__ pkRoot,
    const __bf16* __restrict__ pkIh, const __bf16* __restrict__ pkHh,
    const float* __restrict__ bih, const float* __restrict__ bhh) {
  __shared__ __bf16 m_lds[64 * DDIM];

  int lane = threadIdx.x & 31;
  int wave = threadIdx.x >> 5;
  int blockrow = blockIdx.x * 64;
  int roff = (lane < 16) ? 0 : 8;
  int nlo = lane & 15;

  // ---- phase 1: m = relu(agg@Wrel + out@Wroot + brel), staged to LDS as bf16
#pragma unroll
  for (int t = wave; t < 16; t += 8) {
    int rt = t >> 2, ct = t & 3;
    const float*  aggp = agg + (long)(blockrow + rt * 16) * DDIM;
    const __bf16* outp = xb  + (long)(blockrow + rt * 16) * DDIM;
    v8f acc = {};
#pragma unroll
    for (int k0 = 0; k0 < DDIM; k0 += 32) {
      int kt = k0 >> 5;
      acc = wmma_bf16(loadA_f32(aggp, DDIM, lane, k0),
                      loadB_pk(pkRel, 2, kt, ct, lane), acc);
      acc = wmma_bf16(loadA_gbf(outp, DDIM, lane, k0),
                      loadB_pk(pkRoot, 2, kt, ct, lane), acc);
    }
    int col = ct * 16 + nlo;
    float bv = brel[col];
#pragma unroll
    for (int r = 0; r < 8; ++r) {
      float v = acc[r] + bv;
      v = (v > 0.0f) ? v : 0.0f;
      m_lds[(rt * 16 + r + roff) * DDIM + col] = (__bf16)v;
    }
  }
  __syncthreads();

  // ---- phase 2: GRU gates; gi = m@Wih^T + bih, gh = h@Whh^T + bhh
  // packed W^T has N = 3*DDIM -> nt = gate*4 + ds
#pragma unroll
  for (int t = wave; t < 16; t += 8) {
    int rt = t >> 2, ds = t & 3;
    int rb = blockrow + rt * 16;     // global row base of this tile
    const __bf16* hp0 = xb + (long)rb * DDIM;   // h == xb at layer entry
    v8f ai0 = {}, ai1 = {}, ai2 = {}, ah0 = {}, ah1 = {}, ah2 = {};
#pragma unroll
    for (int k0 = 0; k0 < DDIM; k0 += 32) {
      int kt = k0 >> 5;
      v16bf am = loadA_lds(m_lds, DDIM, rt * 16, lane, k0);
      v16bf ah = loadA_gbf(hp0, DDIM, lane, k0);
      ai0 = wmma_bf16(am, loadB_pk(pkIh, 2, kt, 0 * 4 + ds, lane), ai0);
      ai1 = wmma_bf16(am, loadB_pk(pkIh, 2, kt, 1 * 4 + ds, lane), ai1);
      ai2 = wmma_bf16(am, loadB_pk(pkIh, 2, kt, 2 * 4 + ds, lane), ai2);
      ah0 = wmma_bf16(ah, loadB_pk(pkHh, 2, kt, 0 * 4 + ds, lane), ah0);
      ah1 = wmma_bf16(ah, loadB_pk(pkHh, 2, kt, 1 * 4 + ds, lane), ah1);
      ah2 = wmma_bf16(ah, loadB_pk(pkHh, 2, kt, 2 * 4 + ds, lane), ah2);
    }
    int d = ds * 16 + nlo;
    float bir = bih[d], biz = bih[DDIM + d], bin = bih[2 * DDIM + d];
    float bhr = bhh[d], bhz = bhh[DDIM + d], bhn = bhh[2 * DDIM + d];
#pragma unroll
    for (int r = 0; r < 8; ++r) {
      long idx = (long)(rb + r + roff) * DDIM + d;
      float ir = ai0[r] + bir, iz = ai1[r] + biz, in = ai2[r] + bin;
      float hr = ah0[r] + bhr, hz = ah1[r] + bhz, hn = ah2[r] + bhn;
      float rr = 1.0f / (1.0f + __expf(-(ir + hr)));
      float zz = 1.0f / (1.0f + __expf(-(iz + hz)));
      float nn = tanhf(in + rr * hn);
      float hp = h[idx];                         // f32 GRU state for the blend
      out[idx] = (1.0f - zz) * nn + zz * hp;     // h_new -> out only (no RAW)
    }
  }
  __syncthreads();

  // ---- phase 3: h <- out, xb <- bf16(out) (all reads of h/xb done at barrier)
  long base = (long)blockrow * DDIM;
  for (int i = threadIdx.x; i < 64 * DDIM; i += blockDim.x) {
    float v = out[base + i];
    h[base + i]  = v;
    xb[base + i] = (__bf16)v;
  }
}

// mean-pool accumulate + feats copy
__global__ __launch_bounds__(256) void pool_kernel(const float* __restrict__ out,
                                                   const int* __restrict__ batch,
                                                   float* __restrict__ pool,
                                                   float* __restrict__ feats) {
  long idx = (long)blockIdx.x * blockDim.x + threadIdx.x;
  if (idx >= (long)NNODE * DDIM) return;
  int i = (int)(idx >> 6);
  int d = (int)(idx & 63);
  float v = out[idx];
  feats[idx] = v;
  atomicAdd(&pool[(long)batch[i] * DDIM + d], v);
}

__global__ void finalize_kernel(const float* __restrict__ pool,
                                const float* __restrict__ counts,
                                float* __restrict__ outs) {
  int idx = blockIdx.x * blockDim.x + threadIdx.x;
  if (idx >= GCNT * DDIM) return;
  float c = counts[idx >> 6];
  if (c < 1.0f) c = 1.0f;
  outs[idx] = pool[idx] / c;
}

// ---------------- launch ----------------

extern "C" void kernel_launch(void* const* d_in, const int* in_sizes, int n_in,
                              void* d_out, int out_size, void* d_ws, size_t ws_size,
                              hipStream_t stream) {
  (void)in_sizes; (void)n_in; (void)out_size; (void)ws_size;
  const float* x     = (const float*)d_in[0];
  const int*   ei    = (const int*)d_in[1];
  const float* att   = (const float*)d_in[2];
  const int*   batch = (const int*)d_in[3];
  const float* lin_W = (const float*)d_in[4];
  const float* lin_b = (const float*)d_in[5];
  const float* Wrel  = (const float*)d_in[6];
  const float* brel  = (const float*)d_in[7];
  const float* Wroot = (const float*)d_in[8];
  const float* Wih   = (const float*)d_in[9];
  const float* Whh   = (const float*)d_in[10];
  const float* bih   = (const float*)d_in[11];
  const float* bhh   = (const float*)d_in[12];
  const int* src = ei;
  const int* dst = ei + NEDGE;

  float* outb   = (float*)d_ws;                       // [NPAD, D] f32
  float* hb     = outb + (long)NPAD * DDIM;           // [NPAD, D] f32
  float* aggb   = hb   + (long)NPAD * DDIM;           // [NPAD, D] f32
  float* pool   = aggb + (long)NPAD * DDIM;           // [G, D]
  float* counts = pool + GCNT * DDIM;                 // [G]

  // bf16 regions (32B-aligned: preceding float region is a 32B multiple)
  __bf16* xb      = (__bf16*)(counts + GCNT);                  // [NPAD, D] bf16
  __bf16* xbf     = xb      + (long)NPAD * DDIM;               // [N, F] bf16
  __bf16* pk_lin  = xbf     + (long)NNODE * FDIM;              // NF * F*D
  __bf16* pk_rel  = pk_lin  + (long)NFAC * FDIM * DDIM;        // NF*NL * D*D
  __bf16* pk_root = pk_rel  + (long)NFAC * NLAY * DDIM * DDIM;
  __bf16* pk_ih   = pk_root + (long)NFAC * NLAY * DDIM * DDIM; // NF * D*(3D)
  __bf16* pk_hh   = pk_ih   + (long)NFAC * 3 * DDIM * DDIM;

  float* d_outs  = (float*)d_out;                     // [NF, G, D]
  float* d_feats = d_outs + (long)NFAC * GCNT * DDIM; // [NF, N, D]

  // ---- one-time (per launch) conversions/packing ----
  {
    long nx = (long)NNODE * FDIM;
    cvt_bf16_kernel<<<(int)((nx + 255) / 256), 256, 0, stream>>>(x, xbf, nx);
  }
  for (int f = 0; f < NFAC; ++f) {
    packB_kernel<<<(FDIM * DDIM + 255) / 256, 256, 0, stream>>>(
        lin_W + (long)f * FDIM * DDIM, pk_lin + (long)f * FDIM * DDIM, FDIM, DDIM);
    for (int l = 0; l < NLAY; ++l) {
      long o = ((long)f * NLAY + l) * DDIM * DDIM;
      packB_kernel<<<(DDIM * DDIM + 255) / 256, 256, 0, stream>>>(
          Wrel + o, pk_rel + o, DDIM, DDIM);
      packB_kernel<<<(DDIM * DDIM + 255) / 256, 256, 0, stream>>>(
          Wroot + o, pk_root + o, DDIM, DDIM);
    }
    long og = (long)f * 3 * DDIM * DDIM;
    packBT_kernel<<<(3 * DDIM * DDIM + 255) / 256, 256, 0, stream>>>(
        Wih + og, pk_ih + og, DDIM, 3 * DDIM);
    packBT_kernel<<<(3 * DDIM * DDIM + 255) / 256, 256, 0, stream>>>(
        Whh + og, pk_hh + og, DDIM, 3 * DDIM);
  }

  zero_kernel<<<1, 256, 0, stream>>>(counts, GCNT);
  count_kernel<<<(NNODE + 255) / 256, 256, 0, stream>>>(batch, counts);

  const long na = (long)NPAD * DDIM;
  const long ns = (long)NEDGE * DDIM;
  const long np = (long)NNODE * DDIM;

  for (int f = 0; f < NFAC; ++f) {
    lin_kernel<<<NNODE / 16, 128, 0, stream>>>(
        xbf, pk_lin + (long)f * FDIM * DDIM, lin_b + f * DDIM, outb, hb, xb);
    for (int l = 0; l < NLAY; ++l) {
      long o = ((long)f * NLAY + l) * DDIM * DDIM;
      zero_kernel<<<(int)((na + 255) / 256), 256, 0, stream>>>(aggb, na);
      scatter_kernel<<<(int)((ns + 255) / 256), 256, 0, stream>>>(
          xb, src, dst, att + (long)f * NEDGE, aggb);
      layer_kernel<<<NPAD / 64, 256, 0, stream>>>(
          aggb, outb, hb, xb,
          pk_rel + o, brel + ((long)f * NLAY + l) * DDIM,
          pk_root + o,
          pk_ih + (long)f * 3 * DDIM * DDIM, pk_hh + (long)f * 3 * DDIM * DDIM,
          bih + (long)f * 3 * DDIM, bhh + (long)f * 3 * DDIM);
    }
    zero_kernel<<<(GCNT * DDIM + 255) / 256, 256, 0, stream>>>(pool, GCNT * DDIM);
    pool_kernel<<<(int)((np + 255) / 256), 256, 0, stream>>>(
        outb, batch, pool, d_feats + (long)f * NNODE * DDIM);
    finalize_kernel<<<(GCNT * DDIM + 255) / 256, 256, 0, stream>>>(
        pool, counts, d_outs + (long)f * GCNT * DDIM);
  }
}
